// GraphAttentionLayer_2534030704777
// MI455X (gfx1250) — matrix operand order repaired and checked by reading, not compiled, not run on hardware
//
#include <hip/hip_runtime.h>
#include <hip/hip_bf16.h>

#define GN 4096
#define FIN 256
#define GH 4
#define GDH 64

typedef __attribute__((ext_vector_type(16))) __bf16 v16bf;
typedef __attribute__((ext_vector_type(8)))  float  v8f;

// ---------------------------------------------------------------------------
// Kernel 0: pack adj (0/1 int32) into 1-bit masks: abits[i*(N/32)+w] bit k
// corresponds to adj[i][w*32+k].  Streams the 64MB adj exactly once; the
// attention kernel then reads 2MB of masks instead of 64MB of ints.
// Each thread packs one 32-entry word (8x b128 loads, fully streaming).
// ---------------------------------------------------------------------------
__global__ __launch_bounds__(256) void pack_adj_kernel(
    const int* __restrict__ adj, unsigned* __restrict__ abits)
{
  const size_t w = (size_t)blockIdx.x * 256 + threadIdx.x;   // word index
  const int4* p = (const int4*)(adj + w * 32);
  unsigned mask = 0u;
#pragma unroll
  for (int q = 0; q < 8; ++q) {
    const int4 v = p[q];
    mask |= (v.x != 0 ? 1u : 0u) << (q * 4 + 0);
    mask |= (v.y != 0 ? 1u : 0u) << (q * 4 + 1);
    mask |= (v.z != 0 ? 1u : 0u) << (q * 4 + 2);
    mask |= (v.w != 0 ? 1u : 0u) << (q * 4 + 3);
  }
  abits[w] = mask;
}

// ---------------------------------------------------------------------------
// Kernel 1: h = x @ W per head (f32 FMA, only 0.54 GFLOP), plus
//   s_src = h·a1, s_dst = h·a2, and h stored as bf16 in the WMMA
//   B-fragment (32x16, KxN) lane layout so kernel 2 loads fragments
//   as one contiguous 32B read per lane.
// Fragment layout (ISA 7.12.2, 16-bit B): col N=lane%16, K base = 0 (lane<16)
//   or 8 (lane>=16); e<8 -> K=base+e, e>=8 -> K=base+8+e.  elem offset = lane*16 + e.
// ---------------------------------------------------------------------------
__global__ __launch_bounds__(256) void gat_proj_kernel(
    const float* __restrict__ x, const float* __restrict__ W,
    const float* __restrict__ a, float* __restrict__ s_src,
    float* __restrict__ s_dst, __bf16* __restrict__ h_swz)
{
  __shared__ float xs[4][FIN];
  __shared__ float red1[4][64];
  __shared__ float red2[4][64];
  const int tid   = threadIdx.x;
  const int head  = blockIdx.y;
  const int nbase = blockIdx.x * 4;

  for (int idx = tid; idx < 4 * FIN; idx += 256)
    xs[idx >> 8][idx & 255] = x[(size_t)(nbase + (idx >> 8)) * FIN + (idx & 255)];
  __syncthreads();

  const int ni = tid >> 6;        // node within block (0..3)
  const int d  = tid & 63;        // head-dim
  const int n  = nbase + ni;

  const float* Wp = W + (size_t)head * FIN * GDH + d;
  float acc = 0.f;
#pragma unroll 8
  for (int k = 0; k < FIN; ++k)
    acc = fmaf(xs[ni][k], Wp[(size_t)k * GDH], acc);

  red1[ni][d] = acc * a[head * 2 * GDH + d];
  red2[ni][d] = acc * a[head * 2 * GDH + GDH + d];

  {
    const int jt = n >> 5, kk = n & 31;       // j-tile, K within tile
    const int dt = d >> 4, c  = d & 15;       // d-tile, N within tile
    const int lane = (kk & 8) ? (16 + c) : c;
    const int e    = (kk & 7) + ((kk & 16) ? 8 : 0);
    const size_t idx =
        ((((size_t)(head * (GN / 32) + jt) * 4 + dt) * 32) + lane) * 16 + e;
    h_swz[idx] = (__bf16)acc;
  }
  __syncthreads();

  if (d == 0) {
    float s = 0.f;
#pragma unroll
    for (int t = 0; t < 64; ++t) s += red1[ni][t];
    s_src[head * GN + n] = s;
  } else if (d == 1) {
    float s = 0.f;
#pragma unroll
    for (int t = 0; t < 64; ++t) s += red2[ni][t];
    s_dst[head * GN + n] = s;
  }
}

// ---------------------------------------------------------------------------
// Kernel 1.5: mg[head] = max_j s_dst[head][j].
// Because leaky_relu is monotone, lrelu(s_src[i] + mg) is a provable upper
// bound on every score of row i -> softmax can use a FIXED per-row max,
// killing all streaming-max/rescale work in the attention kernel.
// ---------------------------------------------------------------------------
__global__ __launch_bounds__(256) void smax_kernel(
    const float* __restrict__ s_dst, float* __restrict__ mg)
{
  __shared__ float red[256];
  const int head = blockIdx.x;
  const int t = threadIdx.x;
  float v = -3.0e38f;
  for (int n = t; n < GN; n += 256) v = fmaxf(v, s_dst[head * GN + n]);
  red[t] = v;
  __syncthreads();
  for (int s = 128; s > 0; s >>= 1) {
    if (t < s) red[t] = fmaxf(red[t], red[t + s]);
    __syncthreads();
  }
  if (t == 0) mg[head] = red[0];
}

// ---------------------------------------------------------------------------
// Kernel 2: attention aggregation on bf16 WMMA with fixed softmax max.
// One wave per (16-row i-tile, head); 4 waves/block = 4 heads of the SAME
// i-tile, so the per-row mask words are shared in L1.  Per 32-wide j-tile:
//   1 dword mask + 64B s_dst + 128B pre-swizzled h per lane,
//   16 exp (TRANS, co-executes with WMMA) and 4x v_wmma_f32_16x16x32_bf16.
// No per-tile shuffles, no accumulator rescaling.
// A-fragment layout (ISA 7.12.2, 16-bit A 16x32): row M = lane%16;
//   K base = 0 (lane<16) or 8 (lane>=16); e<8 -> K=base+e, e>=8 -> K=base+8+e.
// C/D layout: VGPR g of lane holds row M = g + (lane<16 ? 0 : 8), col N=lane%16.
// ---------------------------------------------------------------------------
__global__ __launch_bounds__(128) void gat_attn_kernel(
    const unsigned* __restrict__ abits, const float* __restrict__ s_src,
    const float* __restrict__ s_dst, const float* __restrict__ mg,
    const __bf16* __restrict__ h_swz, float* __restrict__ out)
{
  const int head = threadIdx.x >> 5;
  const int lane = threadIdx.x & 31;
  const int i0   = blockIdx.x * 16;
  const int r    = lane & 15;               // my row within the i-tile
  const int i    = i0 + r;
  const int base0 = (lane < 16) ? 0 : 8;    // K base (A frag) == M base (C frag)

  const float si = s_src[head * GN + i];
  const float sb = si + mg[head];
  const float m  = (sb > 0.f) ? sb : 0.01f * sb;   // upper bound on row scores

  float l = 0.f;                                   // per-lane partial denom
  v8f acc0 = {}, acc1 = {}, acc2 = {}, acc3 = {};

  const unsigned* arow = abits + (size_t)i * (GN / 32);
  const float*    drow = s_dst + head * GN;
  const v16bf*    hb   = (const v16bf*)h_swz;

  for (int jt = 0; jt < GN / 32; ++jt) {
    const int j0 = jt * 32;
    __builtin_prefetch(arow + jt + 32, 0, 0);      // global_prefetch_b8

    const unsigned w = arow[jt];                   // 32 mask bits for this tile

    float dv[16];
    *(float4*)(dv + 0)  = *(const float4*)(drow + j0 + base0 + 0);
    *(float4*)(dv + 4)  = *(const float4*)(drow + j0 + base0 + 4);
    *(float4*)(dv + 8)  = *(const float4*)(drow + j0 + base0 + 16);
    *(float4*)(dv + 12) = *(const float4*)(drow + j0 + base0 + 20);

    // P fragment: p = mask ? exp(lrelu(si + s_dst[j]) - m) : 0   (always <= 1)
    v16bf af;
#pragma unroll
    for (int e = 0; e < 16; ++e) {
      const int ke = (e < 8) ? (base0 + e) : (base0 + 8 + e);
      float sv = si + dv[e];
      sv = (sv > 0.f) ? sv : 0.01f * sv;
      float pe = __expf(sv - m);
      pe = ((w >> ke) & 1u) ? pe : 0.f;
      l += pe;
      af[e] = (__bf16)pe;
    }

    // B fragments (pre-swizzled bf16 h) + 4 WMMAs
    const size_t bbase = (size_t)(head * (GN / 32) + jt) * 4 * 32;
    const v16bf b0 = hb[bbase + 0 * 32 + lane];
    const v16bf b1 = hb[bbase + 1 * 32 + lane];
    const v16bf b2 = hb[bbase + 2 * 32 + lane];
    const v16bf b3 = hb[bbase + 3 * 32 + lane];
    acc0 = __builtin_amdgcn_wmma_f32_16x16x32_bf16(false, af, false, b0,
                                                   (short)0, acc0, false, false);
    acc1 = __builtin_amdgcn_wmma_f32_16x16x32_bf16(false, af, false, b1,
                                                   (short)0, acc1, false, false);
    acc2 = __builtin_amdgcn_wmma_f32_16x16x32_bf16(false, af, false, b2,
                                                   (short)0, acc2, false, false);
    acc3 = __builtin_amdgcn_wmma_f32_16x16x32_bf16(false, af, false, b3,
                                                   (short)0, acc3, false, false);
  }

  // merge the two half-lane partial sums of the denominator (once, not per tile)
  l += __shfl_xor(l, 16, 32);

  // finalize: divide by denominator, ELU, store
#pragma unroll
  for (int g = 0; g < 8; ++g) {
    const float lg  = __shfl(l, base0 + g, 32);    // lane M holds row-M denom
    const float inv = (lg > 0.f) ? (1.f / lg) : 0.f;
    const int row   = i0 + base0 + g;
    const size_t ob = (size_t)row * (GH * GDH) + head * GDH + r;
    float v;
    v = acc0[g] * inv; out[ob +  0] = (v > 0.f) ? v : (__expf(v) - 1.f);
    v = acc1[g] * inv; out[ob + 16] = (v > 0.f) ? v : (__expf(v) - 1.f);
    v = acc2[g] * inv; out[ob + 32] = (v > 0.f) ? v : (__expf(v) - 1.f);
    v = acc3[g] * inv; out[ob + 48] = (v > 0.f) ? v : (__expf(v) - 1.f);
  }
}

// ---------------------------------------------------------------------------
extern "C" void kernel_launch(void* const* d_in, const int* in_sizes, int n_in,
                              void* d_out, int out_size, void* d_ws, size_t ws_size,
                              hipStream_t stream) {
  const float* x   = (const float*)d_in[0];   // (4096, 256) f32
  const int*   adj = (const int*)  d_in[1];   // (4096, 4096) i32
  const float* W   = (const float*)d_in[2];   // (4, 256, 64) f32
  const float* a   = (const float*)d_in[3];   // (4, 128, 1) f32
  float* out = (float*)d_out;                 // (4096, 256) f32

  // workspace: s_src | s_dst | mg(pad 16) | h_swz (bf16) | abits (u32)
  float*  s_src = (float*)d_ws;
  float*  s_dst = s_src + GH * GN;
  float*  mg    = s_dst + GH * GN;
  __bf16* h_swz = (__bf16*)(mg + 16);
  unsigned* abits = (unsigned*)(h_swz + (size_t)GH * GN * GDH);

  pack_adj_kernel<<<dim3((GN / 32) * GN / 256), 256, 0, stream>>>(adj, abits);
  gat_proj_kernel<<<dim3(GN / 4, GH), 256, 0, stream>>>(x, W, a, s_src, s_dst, h_swz);
  smax_kernel<<<dim3(GH), 256, 0, stream>>>(s_dst, mg);
  gat_attn_kernel<<<dim3(GN / 16), 128, 0, stream>>>(abits, s_src, s_dst, mg, h_swz, out);
}
